// CrossAttentionModule_bias_82265803588077
// MI455X (gfx1250) — matrix operand order repaired and checked by reading, not compiled, not run on hardware
//
#include <hip/hip_runtime.h>
#include <hip/hip_bf16.h>
#include <math.h>

// ---------------------------------------------------------------------------
// CDNA5 / gfx1250 flash-attention for the CrossAttentionModule reference.
//   B=2, C=256 (head dim), H=W=64 -> N=4096 tokens, fp32 in/out.
//   out[b,:,n] = softmax((Q+qp)(K+kp)^T / 16) @ V   +  c_b * colsum(V)
// Matmuls run on v_wmma_f32_16x16x32_bf16 (f32 accumulate).
// sched_group_barrier pins a DS-read / WMMA software pipeline so fragment
// loads overlap matrix issue instead of s_wait_dscnt 0 before every WMMA.
// ---------------------------------------------------------------------------

typedef __attribute__((ext_vector_type(16))) __bf16 v16bf;
typedef __attribute__((ext_vector_type(8)))  float  v8f;

union BF16x16 {
  v16bf v;
  unsigned short u[16];
};

#define N_TOK 4096
#define CDIM  256
#define KSTR  264   // Ksh row stride (elems): 528 B, 16B aligned, bank-skewed
#define VSTR  40    // Vsh row stride (elems): 80 B, 16B aligned
#define PSTR  17    // P bounce-buffer row stride (elems)

// sched_group_barrier masks
#define SG_WMMA   0x008   // MFMA/WMMA class
#define SG_DSREAD 0x100

__device__ __forceinline__ unsigned short f2bf(float f) {
  unsigned int u = __float_as_uint(f);
  u += 0x7FFFu + ((u >> 16) & 1u);          // round-to-nearest-even
  return (unsigned short)(u >> 16);
}

__device__ __forceinline__ unsigned int pack2bf(float lo, float hi) {
  return (unsigned int)f2bf(lo) | ((unsigned int)f2bf(hi) << 16);
}

__device__ __forceinline__ void ld16(unsigned short* dst, const unsigned short* src) {
  *(uint4*)dst = *(const uint4*)src;        // ds_load_b128 from LDS
}

__device__ __forceinline__ v8f wmma_bf16(const BF16x16& a, const BF16x16& b, v8f c) {
  // (neg_a, A, neg_b, B, c_mod, C, reuse_a, reuse_b)
  return __builtin_amdgcn_wmma_f32_16x16x32_bf16(false, a.v, false, b.v,
                                                 (short)0, c, false, false);
}

__device__ __forceinline__ v8f z8() {
  v8f z = {0.f, 0.f, 0.f, 0.f, 0.f, 0.f, 0.f, 0.f};
  return z;
}

// ---------------------------------------------------------------------------
// Pre-pass: per-batch additive constant c_b and colsum(V)[c] into workspace.
//   ws[0..1]       : c_b
//   ws[2..2+511]   : colsum[b][c]
// ---------------------------------------------------------------------------
__global__ void prep_kernel(const float* __restrict__ values,
                            const float* __restrict__ mask_eye,
                            const float* __restrict__ mask_mouth,
                            const float* __restrict__ bias_eye,
                            const float* __restrict__ bias_mouth,
                            float* __restrict__ ws) {
  const int b = blockIdx.x;
  const int t = threadIdx.x;                 // 256 threads
  __shared__ float red[256];

  // nearest-resize 128x128 -> 64x64 picks mask[2i][2j]; sum of squares
  float se = 0.f, sm = 0.f;
  for (int i = t; i < 4096; i += 256) {
    int r = i >> 6, c = i & 63;
    int idx = b * 16384 + (r << 1) * 128 + (c << 1);
    float e = mask_eye[idx];   se += e * e;
    float m = mask_mouth[idx]; sm += m * m;
  }
  red[t] = se; __syncthreads();
  for (int s = 128; s > 0; s >>= 1) { if (t < s) red[t] += red[t + s]; __syncthreads(); }
  float sume = red[0]; __syncthreads();
  red[t] = sm; __syncthreads();
  for (int s = 128; s > 0; s >>= 1) { if (t < s) red[t] += red[t + s]; __syncthreads(); }
  float summ = red[0];

  if (t == 0) {
    float xe = bias_eye[0] * sume;
    float xm = bias_mouth[0] * summ;
    // numerically stable softplus
    float spe = fmaxf(xe, 0.f) + log1pf(__expf(-fabsf(xe)));
    float spm = fmaxf(xm, 0.f) + log1pf(__expf(-fabsf(xm)));
    ws[b] = spe + spm;
  }

  // colsum(V): one channel per thread, n is contiguous
  float s = 0.f;
  const float* p = values + (size_t)b * CDIM * N_TOK + (size_t)t * N_TOK;
  for (int n = 0; n < N_TOK; ++n) s += p[n];
  ws[2 + b * CDIM + t] = s;
}

// ---------------------------------------------------------------------------
// Flash attention. Block = 128 threads (4 waves). Wave owns 16 query rows.
// Computes S^T = K*Q^T so the softmax reduction is VGPR-local + one shfl,
// and O^T = V^T*P^T so the result lands directly in [B,C,N] layout.
// ---------------------------------------------------------------------------
__global__ void __launch_bounds__(128, 1)
fa_kernel(const float* __restrict__ queries,
          const float* __restrict__ keys,
          const float* __restrict__ values,
          const float* __restrict__ q_pos,
          const float* __restrict__ k_pos,
          const float* __restrict__ ws,
          float* __restrict__ out) {
  __shared__ __align__(16) unsigned short Ksh[32 * KSTR];    // K tile [kv][d]
  __shared__ __align__(16) unsigned short Vsh[256 * VSTR];   // V^T tile [d][kv]
  __shared__ __align__(16) unsigned short Plds[4][32 * PSTR];// per-wave P bounce

  const int tid  = threadIdx.x;
  const int lane = tid & 31;
  const int wave = tid >> 5;
  const int h    = lane >> 4;          // lane half (0/1)
  const int ln   = lane & 15;
  const int b    = blockIdx.x >> 6;
  const int qbase = (blockIdx.x & 63) * 64 + wave * 16;
  const int qg    = qbase + ln;        // this lane's query column

  const size_t bo = (size_t)b * CDIM * N_TOK;
  const float* Qp = queries + bo;
  const float* Kp = keys + bo;
  const float* Vp = values + bo;

  // Persistent Q B-fragments: B(32x16), frag[i] = d = dc*32 + 8h + i + (i&8), col = q
  BF16x16 Qb[8];
#pragma unroll
  for (int dc = 0; dc < 8; ++dc) {
#pragma unroll
    for (int i = 0; i < 16; ++i) {
      int d = dc * 32 + 8 * h + i + (i & 8);
      float q = Qp[(size_t)d * N_TOK + qg] + q_pos[qg * CDIM + d];
      Qb[dc].u[i] = f2bf(q * 0.0625f);   // fold SCALE=1/16 (exact in bf16)
    }
  }

  v8f acc[16];                           // O^T accumulator: 16 tiles of 16d x 16q
#pragma unroll
  for (int t = 0; t < 16; ++t) acc[t] = z8();
  float mrun = -3.0e38f;
  float lrun = 0.f;

  for (int j = 0; j < 128; ++j) {        // 32-wide kv tiles
    const int jb = j << 5;
    __syncthreads();                     // protect LDS tiles from prev iter

    if (j < 127) {                       // global_prefetch_b8 of next tile
      __builtin_prefetch(Kp + (size_t)lane * N_TOK + jb + 32, 0, 0);
      __builtin_prefetch(Vp + (size_t)tid  * N_TOK + jb + 32, 0, 0);
    }

    // K tile -> Ksh[kv][d]. Pass p: wave reads one coalesced 128B row of keys
    // per d (all lanes same d, kv = lane-contiguous n); k_pos as float4;
    // 4 packed bf16 go out as one ds_store_b64.
    {
      const int kv = tid & 31;
      const int dq = (tid >> 5) << 2;            // this wave's 4-d subgroup
      const float* kpr  = k_pos + (jb + kv) * CDIM;
      const float* krow = Kp + jb + kv;
#pragma unroll 4
      for (int p = 0; p < 16; ++p) {
        int d0 = p * 16 + dq;
        float4 kp4 = *(const float4*)(kpr + d0);
        float e0 = krow[(size_t)(d0 + 0) * N_TOK] + kp4.x;
        float e1 = krow[(size_t)(d0 + 1) * N_TOK] + kp4.y;
        float e2 = krow[(size_t)(d0 + 2) * N_TOK] + kp4.z;
        float e3 = krow[(size_t)(d0 + 3) * N_TOK] + kp4.w;
        uint2 w;
        w.x = pack2bf(e0, e1);
        w.y = pack2bf(e2, e3);
        *(uint2*)&Ksh[kv * KSTR + d0] = w;
      }
    }
    // V tile -> Vsh[d][kv]; float4 global reads, packed b128 LDS stores
#pragma unroll
    for (int r = 0; r < 2; ++r) {
      int d = tid + (r << 7);
      const float4* s4 = (const float4*)(Vp + (size_t)d * N_TOK + jb);
      unsigned short* dst = &Vsh[d * VSTR];
#pragma unroll
      for (int c = 0; c < 4; ++c) {
        float4 a  = s4[2 * c];
        float4 bb = s4[2 * c + 1];
        uint4 o;
        o.x = pack2bf(a.x,  a.y);
        o.y = pack2bf(a.z,  a.w);
        o.z = pack2bf(bb.x, bb.y);
        o.w = pack2bf(bb.z, bb.w);
        *(uint4*)(dst + c * 8) = o;
      }
    }
    __syncthreads();

    // S^T = K * Q^T : two 16(kv)x16(q) tiles, K-depth 256 in 8 chunks of 32.
    v8f s0 = z8(), s1 = z8();
    BF16x16 ka0[2], ka1[2];
    auto loadK = [&](int dc, int buf) {
      const unsigned short* p0 = &Ksh[ln * KSTR + dc * 32 + 8 * h];
      ld16(&ka0[buf].u[0], p0);
      ld16(&ka0[buf].u[8], p0 + 16);
      const unsigned short* p1 = &Ksh[(16 + ln) * KSTR + dc * 32 + 8 * h];
      ld16(&ka1[buf].u[0], p1);
      ld16(&ka1[buf].u[8], p1 + 16);
    };
    loadK(0, 0);
#pragma unroll
    for (int dc = 0; dc < 8; ++dc) {
      const int cur = dc & 1;
      if (dc < 7) loadK(dc + 1, cur ^ 1);
      s0 = wmma_bf16(ka0[cur], Qb[dc], s0);
      s1 = wmma_bf16(ka1[cur], Qb[dc], s1);
    }
    // Pin pipeline: [DSr x4] then 7x[DSr x4, WMMA x2] then [WMMA x2]:
    // 8 DS reads outstanding before the first WMMA -> staggered dscnt waits.
    __builtin_amdgcn_sched_group_barrier(SG_DSREAD, 4, 0);
#pragma unroll
    for (int it = 0; it < 7; ++it) {
      __builtin_amdgcn_sched_group_barrier(SG_DSREAD, 4, 0);
      __builtin_amdgcn_sched_group_barrier(SG_WMMA,   2, 0);
    }
    __builtin_amdgcn_sched_group_barrier(SG_WMMA, 2, 0);

    // Online softmax over kv (lane's fixed q column): 16 local + shfl_xor(16)
    float tmax = -3.0e38f;
#pragma unroll
    for (int g = 0; g < 8; ++g) tmax = fmaxf(tmax, fmaxf(s0[g], s1[g]));
    tmax = fmaxf(tmax, __shfl_xor(tmax, 16, 32));
    float mnew  = fmaxf(mrun, tmax);
    float alpha = __expf(mrun - mnew);   // first iter: exp(-inf)=0
    float psum  = 0.f;
    unsigned short* pl = &Plds[wave][0];
#pragma unroll
    for (int g = 0; g < 8; ++g) {
      float p0v = __expf(s0[g] - mnew);
      float p1v = __expf(s1[g] - mnew);
      psum += p0v + p1v;
      pl[(g + 8 * h) * PSTR + ln]      = f2bf(p0v);   // kv = g+8h
      pl[(16 + g + 8 * h) * PSTR + ln] = f2bf(p1v);   // kv = 16+g+8h
    }
    psum += __shfl_xor(psum, 16, 32);
    lrun = lrun * alpha + psum;
    mrun = mnew;
#pragma unroll
    for (int t = 0; t < 16; ++t) acc[t] *= alpha;

    // P^T as B-fragment (C-layout -> B-layout lane transpose via LDS bounce)
    BF16x16 pb;
#pragma unroll
    for (int i = 0; i < 16; ++i) pb.u[i] = pl[(8 * h + i + (i & 8)) * PSTR + ln];

    // O^T += V^T * P^T : A(16x32) = V^T[dtile][kv]; double-buffered A frags
    BF16x16 av[2];
    auto loadV = [&](int dt, int buf) {
      const unsigned short* vp = &Vsh[(dt * 16 + ln) * VSTR + 8 * h];
      ld16(&av[buf].u[0], vp);
      ld16(&av[buf].u[8], vp + 16);
    };
    loadV(0, 0);
#pragma unroll
    for (int dt = 0; dt < 16; ++dt) {
      const int cur = dt & 1;
      if (dt < 15) loadV(dt + 1, cur ^ 1);
      acc[dt] = wmma_bf16(av[cur], pb, acc[dt]);
    }
    // Pin pipeline: pb's 16 u16 reads first, then [DSr x2] ahead of each WMMA.
    __builtin_amdgcn_sched_group_barrier(SG_DSREAD, 16, 0);
    __builtin_amdgcn_sched_group_barrier(SG_DSREAD, 2, 0);
#pragma unroll
    for (int it = 0; it < 15; ++it) {
      __builtin_amdgcn_sched_group_barrier(SG_DSREAD, 2, 0);
      __builtin_amdgcn_sched_group_barrier(SG_WMMA,   1, 0);
    }
    __builtin_amdgcn_sched_group_barrier(SG_WMMA, 1, 0);
  }

  // Epilogue: normalize + analytic mask-bias term; O^T is already [C][N]
  const float  invl   = 1.f / lrun;
  const float  cb     = ws[b];
  const float* colsum = ws + 2 + b * CDIM;
  float* outp = out + bo;
#pragma unroll
  for (int dt = 0; dt < 16; ++dt) {
#pragma unroll
    for (int g = 0; g < 8; ++g) {
      int d = dt * 16 + g + 8 * h;       // C-layout: M = g + 8*half
      outp[(size_t)d * N_TOK + qg] = acc[dt][g] * invl + cb * colsum[d];
    }
  }
}

extern "C" void kernel_launch(void* const* d_in, const int* in_sizes, int n_in,
                              void* d_out, int out_size, void* d_ws, size_t ws_size,
                              hipStream_t stream) {
  const float* queries    = (const float*)d_in[0];
  const float* keys       = (const float*)d_in[1];
  const float* values     = (const float*)d_in[2];
  const float* mask_eye   = (const float*)d_in[3];
  const float* mask_mouth = (const float*)d_in[4];
  const float* q_pos      = (const float*)d_in[5];
  const float* k_pos      = (const float*)d_in[6];
  const float* bias_eye   = (const float*)d_in[7];
  const float* bias_mouth = (const float*)d_in[8];
  float* ws  = (float*)d_ws;
  float* out = (float*)d_out;

  prep_kernel<<<dim3(2), dim3(256), 0, stream>>>(values, mask_eye, mask_mouth,
                                                 bias_eye, bias_mouth, ws);
  // B=2 batches x 64 query-blocks (64 rows each) = 128 blocks, 4 waves/block
  fa_kernel<<<dim3(128), dim3(128), 0, stream>>>(queries, keys, values,
                                                 q_pos, k_pos, ws, out);
}
